// KVQ_self_attention_63376537420541
// MI455X (gfx1250) — matrix operand order repaired and checked by reading, not compiled, not run on hardware
//
#include <hip/hip_runtime.h>
#include <math.h>
#include <stdint.h>

#define N_TOK 8192
#define DMODEL 256

typedef __attribute__((ext_vector_type(16))) _Float16 v16h;
typedef __attribute__((ext_vector_type(8)))  _Float16 v8h;
typedef __attribute__((ext_vector_type(8)))  float    v8f;
typedef __attribute__((ext_vector_type(4)))  float    v4f;

// ---- WMMA helpers (CDNA5 16x16x32 f16 -> f32) --------------------------------
__device__ __forceinline__ v8f wmma_f16(v16h a, v16h b, v8f c) {
  // 8 args: (neg_a, A, neg_b, B, c_mod, C, reuse_a, reuse_b)
  return __builtin_amdgcn_wmma_f32_16x16x32_f16(false, a, false, b, (short)0, c,
                                                false, false);
}

// A-matrix fragment (16x32 f16): lane l holds row m=l%16,
// k-range 16h..16h+15, h = l/16.
__device__ __forceinline__ v16h load_afrag(const _Float16* row, int hf) {
  v8h lo = *(const v8h*)(row + hf * 16);
  v8h hi = *(const v8h*)(row + hf * 16 + 8);
  return __builtin_shufflevector(lo, hi, 0, 1, 2, 3, 4, 5, 6, 7, 8, 9, 10, 11,
                                 12, 13, 14, 15);
}

// B-matrix fragment (32x16 f16), sourced from B^T stored row-major (k contiguous):
// lane l holds column n=l%16, k-range 16h..16h+15.
__device__ __forceinline__ v16h load_bfrag(const _Float16* rowT, int hf) {
  return *(const v16h*)(rowT + hf * 16);
}

// CDNA5 async global->LDS copy (16 bytes/lane), tracked with ASYNCcnt.
__device__ __forceinline__ void async_copy_b128(uint32_t lds_off,
                                                const void* gaddr) {
  asm volatile("global_load_async_to_lds_b128 %0, %1, off" ::"v"(lds_off),
               "v"(gaddr)
               : "memory");
}

// ---- Kernel 1: fused QKV projection (f32 in -> f16 out) ----------------------
// grid (N/64, D/64, 3), block 128 (4 waves). which=0:Q, 1:K, 2:V(transposed).
__global__ __launch_bounds__(128) void qkv_proj_kernel(
    const float* __restrict__ X, const float* __restrict__ WQ,
    const float* __restrict__ WK, const float* __restrict__ WV,
    _Float16* __restrict__ Qh, _Float16* __restrict__ Kh,
    _Float16* __restrict__ Vt) {
  __shared__ __attribute__((aligned(16))) _Float16 Xl[64][32];
  __shared__ __attribute__((aligned(16))) _Float16 Wt[64][32];  // Wt[n][k]

  const int t    = threadIdx.x;
  const int lane = t & 31;
  const int wid  = t >> 5;      // wave 0..3
  const int hf   = lane >> 4;   // half-wave 0/1
  const int l16  = lane & 15;

  const int mbase = blockIdx.x * 64;
  const int nbase = blockIdx.y * 64;
  const int which = blockIdx.z;
  const float* W  = (which == 0) ? WQ : ((which == 1) ? WK : WV);

  const int wm = (wid >> 1) * 32;  // wave's 32 rows within tile
  const int wn = (wid & 1) * 32;   // wave's 32 cols within tile

  v8f acc[2][2] = {};

  for (int kb = 0; kb < DMODEL; kb += 32) {
    __syncthreads();
    // X tile [64][32] f32 -> f16 (vectorized f32x4 reads)
#pragma unroll
    for (int i = 0; i < 4; ++i) {
      int e  = t + i * 128;      // 512 chunks of 4 floats
      int r  = e >> 3;
      int c4 = (e & 7) * 4;
      v4f xv = *(const v4f*)(X + (size_t)(mbase + r) * DMODEL + kb + c4);
      Xl[r][c4 + 0] = (_Float16)xv.x;
      Xl[r][c4 + 1] = (_Float16)xv.y;
      Xl[r][c4 + 2] = (_Float16)xv.z;
      Xl[r][c4 + 3] = (_Float16)xv.w;
    }
    // W tile transposed: Wt[n][k] = W[kb+k][nbase+n] (coalesced over n)
#pragma unroll
    for (int i = 0; i < 16; ++i) {
      int e = t + i * 128;       // 2048 elements
      int k = e >> 6;
      int n = e & 63;
      Wt[n][k] = (_Float16)W[(size_t)(kb + k) * DMODEL + nbase + n];
    }
    __syncthreads();

#pragma unroll
    for (int mt = 0; mt < 2; ++mt) {
      v16h a = load_afrag(&Xl[wm + mt * 16 + l16][0], hf);
#pragma unroll
      for (int nt = 0; nt < 2; ++nt) {
        v16h b = load_bfrag(&Wt[wn + nt * 16 + l16][0], hf);
        acc[mt][nt] = wmma_f16(a, b, acc[mt][nt]);
      }
    }
  }

  // store f16: Q/K row-major [N][D]; V transposed [D][N]
#pragma unroll
  for (int mt = 0; mt < 2; ++mt)
#pragma unroll
    for (int nt = 0; nt < 2; ++nt)
#pragma unroll
      for (int r = 0; r < 8; ++r) {
        int row = mbase + wm + mt * 16 + r + hf * 8;
        int col = nbase + wn + nt * 16 + l16;
        _Float16 hv = (_Float16)acc[mt][nt][r];
        if (which == 0)
          Qh[(size_t)row * DMODEL + col] = hv;
        else if (which == 1)
          Kh[(size_t)row * DMODEL + col] = hv;
        else
          Vt[(size_t)col * N_TOK + row] = hv;
      }
}

// ---- Kernel 2: flash attention with async-LDS pipeline -----------------------
// grid N/64 blocks, 128 threads (4 waves). Wave w owns query rows [64b+16w, +16),
// full 256-column f32 accumulator in 16 C-fragments. K tiles are double-buffered
// and prefetched a full iteration ahead via GLOBAL_LOAD_ASYNC_TO_LDS_B128; the
// V tile load is hidden under S-compute + softmax.
__global__ __launch_bounds__(128) void attention_kernel(
    const _Float16* __restrict__ Qh, const _Float16* __restrict__ Kh,
    const _Float16* __restrict__ Vt, float* __restrict__ Out) {
  __shared__ __attribute__((aligned(16))) _Float16 Kl[2][32][DMODEL];  // ping-pong
  __shared__ __attribute__((aligned(16))) _Float16 Vl[DMODEL][32];     // V^T tile
  __shared__ __attribute__((aligned(16))) _Float16 Pl[4][16][32];      // per-wave P

  const int t    = threadIdx.x;
  const int lane = t & 31;
  const int wid  = t >> 5;
  const int hf   = lane >> 4;
  const int l16  = lane & 15;

  const int qbase = blockIdx.x * 64;
  const int qrow  = qbase + wid * 16 + l16;  // this lane's A-matrix row

  const int JBLK = N_TOK / 32;  // 256 KV steps

  // Issue this thread's 8 async 16B chunks of the K tile for block jb.
  auto issueK = [&](int jb, int buf) {
#pragma unroll
    for (int i = 0; i < 8; ++i) {
      int e = t + i * 128;  // 1024 chunks of 8 halves
      int r = e >> 5;
      int c = (e & 31) * 8;
      uint32_t lds = (uint32_t)(uintptr_t)&Kl[buf][r][c];
      async_copy_b128(lds, (const void*)(Kh + (size_t)(jb * 32 + r) * DMODEL + c));
    }
  };
  // Issue this thread's 8 async 16B chunks of the V^T tile for block jb.
  auto issueV = [&](int jb) {
#pragma unroll
    for (int i = 0; i < 8; ++i) {
      int e = t + i * 128;
      int d = e >> 2;
      int c = (e & 3) * 8;
      uint32_t lds = (uint32_t)(uintptr_t)&Vl[d][c];
      async_copy_b128(lds, (const void*)(Vt + (size_t)d * N_TOK + jb * 32 + c));
    }
  };

  // Q A-fragments held in registers for the whole KV sweep (8 x v16h)
  v16h aq[8];
#pragma unroll
  for (int ks = 0; ks < 8; ++ks)
    aq[ks] = load_afrag(Qh + (size_t)qrow * DMODEL + ks * 32, hf);

  v8f acc[16] = {};
  float mrow[8], lrow[8];
#pragma unroll
  for (int r = 0; r < 8; ++r) {
    mrow[r] = -__builtin_inff();
    lrow[r] = 0.0f;
  }

  issueK(0, 0);  // preamble prefetch: ASYNCcnt = 8

  for (int jb = 0; jb < JBLK; ++jb) {
    const int buf = jb & 1;
    // Drain K[jb] (only it is outstanding here).
    asm volatile("s_wait_asynccnt 0x0" ::: "memory");
    __syncthreads();  // K[jb] globally visible; everyone done with V[jb-1] + K[jb+1]'s buffer

    // Prefetch: V[jb] now (its buffer is free), then K[jb+1] a full step ahead.
    issueV(jb);                            // +8  (oldest outstanding)
    issueK((jb + 1) % JBLK, buf ^ 1);      // +8  (wrapped dummy on last iter keeps counts uniform)

    // S = Q K^T : 16 rows x 32 keys, K-loop over D=256 (8 wmma k-steps x 2 tiles)
    v8f s0 = {}, s1 = {};
#pragma unroll
    for (int ks = 0; ks < 8; ++ks) {
      v16h b0 = load_bfrag(&Kl[buf][l16][ks * 32], hf);
      v16h b1 = load_bfrag(&Kl[buf][16 + l16][ks * 32], hf);
      s0 = wmma_f16(aq[ks], b0, s0);
      s1 = wmma_f16(aq[ks], b1, s1);
    }

    // online softmax: each row m = r + 8*hf lives in one 16-lane half
#pragma unroll
    for (int r = 0; r < 8; ++r) {
      float mx = fmaxf(s0[r], s1[r]);
      mx = fmaxf(mx, __shfl_xor(mx, 1));
      mx = fmaxf(mx, __shfl_xor(mx, 2));
      mx = fmaxf(mx, __shfl_xor(mx, 4));
      mx = fmaxf(mx, __shfl_xor(mx, 8));
      float mnew = fmaxf(mrow[r], mx);
      float p0 = __expf(s0[r] - mnew);
      float p1 = __expf(s1[r] - mnew);
      float rs = p0 + p1;
      rs += __shfl_xor(rs, 1);
      rs += __shfl_xor(rs, 2);
      rs += __shfl_xor(rs, 4);
      rs += __shfl_xor(rs, 8);
      float alpha = __expf(mrow[r] - mnew);  // 0 on first hit (m=-inf)
      lrow[r] = lrow[r] * alpha + rs;
      mrow[r] = mnew;
#pragma unroll
      for (int nt = 0; nt < 16; ++nt) acc[nt][r] *= alpha;
      int m = r + hf * 8;
      Pl[wid][m][l16]      = (_Float16)p0;
      Pl[wid][m][l16 + 16] = (_Float16)p1;
    }
    // intra-wave LDS RAW on P: DS ops are in-order per wave; force wait
    asm volatile("s_wait_dscnt 0" ::: "memory");
    v16h pa = load_afrag(&Pl[wid][l16][0], hf);

    // Drain V[jb] (K[jb+1]'s 8 stay in flight), then make it block-visible.
    asm volatile("s_wait_asynccnt 0x8" ::: "memory");
    __syncthreads();

    // P (16x32) @ V (32x256): one wmma k-step per output tile
#pragma unroll
    for (int nt = 0; nt < 16; ++nt) {
      v16h bv = load_bfrag(&Vl[nt * 16 + l16][0], hf);
      acc[nt] = wmma_f16(pa, bv, acc[nt]);
    }
  }

  // epilogue: O = acc / l, f32 out
#pragma unroll
  for (int r = 0; r < 8; ++r) {
    int row   = qbase + wid * 16 + r + hf * 8;
    float inv = 1.0f / lrow[r];
#pragma unroll
    for (int nt = 0; nt < 16; ++nt)
      Out[(size_t)row * DMODEL + nt * 16 + l16] = acc[nt][r] * inv;
  }
}

// ---- Host launch -------------------------------------------------------------
extern "C" void kernel_launch(void* const* d_in, const int* in_sizes, int n_in,
                              void* d_out, int out_size, void* d_ws,
                              size_t ws_size, hipStream_t stream) {
  (void)in_sizes; (void)n_in; (void)out_size; (void)ws_size;
  const float* X  = (const float*)d_in[0];
  const float* WQ = (const float*)d_in[1];
  const float* WK = (const float*)d_in[2];
  const float* WV = (const float*)d_in[3];
  float* Out = (float*)d_out;

  char* ws = (char*)d_ws;
  const size_t qkv_bytes = (size_t)N_TOK * DMODEL * sizeof(_Float16);  // 4 MB
  _Float16* Qh = (_Float16*)(ws);
  _Float16* Kh = (_Float16*)(ws + qkv_bytes);
  _Float16* Vt = (_Float16*)(ws + 2 * qkv_bytes);

  qkv_proj_kernel<<<dim3(N_TOK / 64, DMODEL / 64, 3), 128, 0, stream>>>(
      X, WQ, WK, WV, Qh, Kh, Vt);
  attention_kernel<<<dim3(N_TOK / 64), 128, 0, stream>>>(Qh, Kh, Vt, Out);
}
